// ElmanNetwork_37168646980318
// MI455X (gfx1250) — compile-verified, matrix operand order checked by reading, not run on hardware
//
#include <hip/hip_runtime.h>
#include <hip/hip_bf16.h>
#include <stdint.h>

// ---------------------------------------------------------------------------
// Elman RNN for MI455X (gfx1250, wave32, WMMA).
//   xh = x @ W_x2h^T + b_x2h           (tiled bf16 WMMA GEMM)
//   for t: h = relu(xh[:,t,:] + h @ W_h2h^T)    (512 step kernels, W in L2)
//   y  = h_T @ W_h2y^T + b_h2y         (tiled bf16 WMMA GEMM, fp32 out)
//
// One templated GEMM body for all three phases:
//  - block = 256 threads (8 waves), computes 64 rows x 256 cols
//  - A chunk (64x32 bf16 = 4 KB) staged in LDS, double-buffered, shared by
//    all 8 waves (8x cut in A global traffic)
//  - each wave computes 4Mx2N 16x16 tiles: 8 WMMAs per K-chunk, A fragment
//    reused across 2 N tiles, B rows read once per 4 M tiles
// ---------------------------------------------------------------------------

typedef __attribute__((ext_vector_type(16))) __bf16 v16bf;
typedef __attribute__((ext_vector_type(8)))  __bf16 v8bf;
typedef __attribute__((ext_vector_type(8)))  float  v8f;

#define B_SZ  64
#define T_SZ  512
#define D_IN  1024
#define D_H   2048
#define D_OUT 1024

// ---- fp32 -> bf16 with round-to-nearest-even -------------------------------
__device__ __forceinline__ uint16_t f2bf(float f) {
  uint32_t u = __float_as_uint(f);
  u += 0x7FFFu + ((u >> 16) & 1u);
  return (uint16_t)(u >> 16);
}

__global__ void __launch_bounds__(256)
convert_f32_to_bf16(const float* __restrict__ src, uint16_t* __restrict__ dst, int n) {
  int i = blockIdx.x * blockDim.x + threadIdx.x;
  if (i < n) dst[i] = f2bf(src[i]);
}

__global__ void __launch_bounds__(256)
zero_u32(uint32_t* __restrict__ p, int n) {
  int i = blockIdx.x * blockDim.x + threadIdx.x;
  if (i < n) p[i] = 0u;
}

// ---------------------------------------------------------------------------
// Unified tiled WMMA GEMM:  out[m][n] = sum_k A[m][k] * Bw[n][k]  (+ epilogue)
//   A  : row-major [rows x K] bf16
//   Bw : row-major [NTOT x K] bf16 (i.e. W, so the GEMM computes A @ W^T)
// MODE 0: outF[m][n] = acc + bias[n]                       (fp32 store)
// MODE 1: outH[m][n] = bf16(relu(acc + xh[m][t][n]))       (recurrence step)
//
// grid.x = (rows/64) * (NTOT/256); block = 256 threads.
// A fragment layout per ISA 7.12.2 (16-bit A 16x32): lane holds row lane&15,
// K chunks [kb,kb+7] and [kb+16,kb+23], kb = (lane>>4)*8.
// ---------------------------------------------------------------------------
template<int K, int NTOT, int MODE>
__global__ void __launch_bounds__(256)
wmma_gemm(const uint16_t* __restrict__ A, const uint16_t* __restrict__ Bw,
          const float* __restrict__ bias, const float* __restrict__ xh,
          int t, float* __restrict__ outF, uint16_t* __restrict__ outH) {
  constexpr int COL_STRIPS = NTOT / 256;
  __shared__ uint16_t smA[2][64 * 32];   // double-buffered A chunk, 8 KB

  const int lane = threadIdx.x & 31;
  const int wave = threadIdx.x >> 5;
  const int rowGroup = blockIdx.x / COL_STRIPS;
  const int colStrip = blockIdx.x % COL_STRIPS;
  const int rowBase  = rowGroup * 64;
  const int n0 = colStrip * 256 + wave * 32;   // wave's two N tiles: n0, n0+16

  // A-chunk stager: each thread moves one 16B slab (row, 8 k's) per chunk
  const int ldRow = threadIdx.x >> 2;          // 0..63
  const int ldKp  = (threadIdx.x & 3) << 3;    // 0,8,16,24
  const uint16_t* aRowPtr = A + (size_t)(rowBase + ldRow) * K + ldKp;

  // B row pointers for the two N tiles (row-major weights [N][K])
  const uint16_t* bRow0 = Bw + (size_t)(n0 + (lane & 15)) * K;
  const uint16_t* bRow1 = Bw + (size_t)(n0 + 16 + (lane & 15)) * K;
  const int kb = (lane >> 4) << 3;

  v8f acc[4][2] = {};

  // preload chunk 0 into buffer 0
  {
    uint4 d = *reinterpret_cast<const uint4*>(aRowPtr);
    *reinterpret_cast<uint4*>(&smA[0][ldRow * 32 + ldKp]) = d;
  }
  __syncthreads();

  for (int kc = 0; kc < K; kc += 32) {
    const int buf = (kc >> 5) & 1;

    // stage next A chunk into the other buffer (global -> LDS)
    if (kc + 32 < K) {
      uint4 d = *reinterpret_cast<const uint4*>(aRowPtr + kc + 32);
      *reinterpret_cast<uint4*>(&smA[buf ^ 1][ldRow * 32 + ldKp]) = d;
    }

    // B fragments straight from global (K contiguous, 2x16B per lane each)
    v16bf bfr[2];
    {
      const uint16_t* p0 = bRow0 + kc + kb;
      const uint16_t* p1 = bRow1 + kc + kb;
      v8bf l0 = *(const v8bf*)p0, h0 = *(const v8bf*)(p0 + 16);
      v8bf l1 = *(const v8bf*)p1, h1 = *(const v8bf*)(p1 + 16);
#pragma unroll
      for (int i = 0; i < 8; ++i) {
        bfr[0][i] = l0[i]; bfr[0][i + 8] = h0[i];
        bfr[1][i] = l1[i]; bfr[1][i + 8] = h1[i];
      }
      if (kc + 64 < K) {
        __builtin_prefetch(p0 + 64, 0, 0);
        __builtin_prefetch(p1 + 64, 0, 0);
      }
    }

    // 4 A fragments from LDS, each feeding 2 WMMAs
#pragma unroll
    for (int mt = 0; mt < 4; ++mt) {
      const uint16_t* ap = &smA[buf][(mt * 16 + (lane & 15)) * 32 + kb];
      v8bf lo = *(const v8bf*)ap;
      v8bf hi = *(const v8bf*)(ap + 16);
      v16bf af;
#pragma unroll
      for (int i = 0; i < 8; ++i) { af[i] = lo[i]; af[i + 8] = hi[i]; }
      acc[mt][0] = __builtin_amdgcn_wmma_f32_16x16x32_bf16(
          false, af, false, bfr[0], (short)0, acc[mt][0], false, false);
      acc[mt][1] = __builtin_amdgcn_wmma_f32_16x16x32_bf16(
          false, af, false, bfr[1], (short)0, acc[mt][1], false, false);
    }
    __syncthreads();
  }

  // epilogue (C/D layout: lane n = lane&15, VGPR v -> m = (lane>>4)*8 + v)
#pragma unroll
  for (int mt = 0; mt < 4; ++mt) {
#pragma unroll
    for (int nt = 0; nt < 2; ++nt) {
      const int n  = n0 + nt * 16 + (lane & 15);
      const int m0 = rowBase + mt * 16 + ((lane >> 4) << 3);
#pragma unroll
      for (int v = 0; v < 8; ++v) {
        if constexpr (MODE == 0) {
          outF[(size_t)(m0 + v) * NTOT + n] = acc[mt][nt][v] + bias[n];
        } else {
          const int bi = m0 + v;   // batch index (rows == B)
          float xv = acc[mt][nt][v] + xh[((size_t)bi * T_SZ + t) * D_H + n];
          xv = xv > 0.f ? xv : 0.f;
          outH[(size_t)bi * D_H + n] = f2bf(xv);
        }
      }
    }
  }
}

// ---------------------------------------------------------------------------
extern "C" void kernel_launch(void* const* d_in, const int* in_sizes, int n_in,
                              void* d_out, int out_size, void* d_ws, size_t ws_size,
                              hipStream_t stream) {
  (void)in_sizes; (void)n_in; (void)out_size; (void)ws_size;

  const float* x      = (const float*)d_in[0];  // [B,T,D_IN]
  const float* W_x2h  = (const float*)d_in[1];  // [D_H, D_IN]
  const float* b_x2h  = (const float*)d_in[2];  // [D_H]
  const float* W_h2h  = (const float*)d_in[3];  // [D_H, D_H]
  const float* W_h2y  = (const float*)d_in[4];  // [D_OUT, D_H]
  const float* b_h2y  = (const float*)d_in[5];  // [D_OUT]
  float* y = (float*)d_out;                     // [B, D_OUT]

  // Workspace layout (bytes), all 16B-aligned:
  char* ws = (char*)d_ws;
  uint16_t* xb  = (uint16_t*)(ws);                         //  64 MB bf16 x
  uint16_t* wxb = (uint16_t*)(ws + 67108864);              //   4 MB bf16 W_x2h
  uint16_t* whb = (uint16_t*)(ws + 71303168);              //   8 MB bf16 W_h2h
  uint16_t* wyb = (uint16_t*)(ws + 79691776);              //   4 MB bf16 W_h2y
  float*    xh  = (float*)(ws + 83886080);                 // 256 MB f32 xh
  uint16_t* h0  = (uint16_t*)(ws + 352321536);             // 256 KB h buf A
  uint16_t* h1  = (uint16_t*)(ws + 352583680);             // 256 KB h buf B

  // 1) fp32 -> bf16 conversions
  {
    int nx = B_SZ * T_SZ * D_IN;
    convert_f32_to_bf16<<<(nx + 255) / 256, 256, 0, stream>>>(x, xb, nx);
    int nwx = D_H * D_IN;
    convert_f32_to_bf16<<<(nwx + 255) / 256, 256, 0, stream>>>(W_x2h, wxb, nwx);
    int nwh = D_H * D_H;
    convert_f32_to_bf16<<<(nwh + 255) / 256, 256, 0, stream>>>(W_h2h, whb, nwh);
    int nwy = D_OUT * D_H;
    convert_f32_to_bf16<<<(nwy + 255) / 256, 256, 0, stream>>>(W_h2y, wyb, nwy);
  }

  // 2) h0 = 0  (B*D_H bf16 == 65536 dwords)
  {
    int nz = (B_SZ * D_H) / 2;
    zero_u32<<<(nz + 255) / 256, 256, 0, stream>>>((uint32_t*)h0, nz);
  }

  // 3) xh = x @ W_x2h^T + b : (32768/64 rowGroups) * (2048/256 strips) = 4096
  wmma_gemm<D_IN, D_H, 0><<<4096, 256, 0, stream>>>(
      xb, wxb, b_x2h, nullptr, 0, xh, nullptr);

  // 4) recurrence: 512 step kernels, double-buffered h; each step reads bf16
  //    W_h2h (8 MB, L2-resident) exactly once. grid = 1 * (2048/256) = 8.
  uint16_t* hb[2] = { h0, h1 };
  for (int t = 0; t < T_SZ; ++t) {
    wmma_gemm<D_H, D_H, 1><<<8, 256, 0, stream>>>(
        hb[t & 1], whb, nullptr, xh, t, nullptr, hb[(t + 1) & 1]);
  }

  // 5) y = h_T @ W_h2y^T + b : grid = 1 * (1024/256) = 4
  wmma_gemm<D_H, D_OUT, 0><<<4, 256, 0, stream>>>(
      hb[T_SZ & 1], wyb, b_h2y, nullptr, 0, y, nullptr);
}
